// MolecGN_63402307223703
// MI455X (gfx1250) — matrix operand order repaired and checked by reading, not compiled, not run on hardware
//
#include <hip/hip_runtime.h>
#include <hip/hip_bf16.h>

// ---------------------------------------------------------------------------
// Fused 2-layer molecular GraphNet for MI455X (gfx1250, wave32, WMMA).
// One 256-thread block (8 waves) per molecule; all intermediates in LDS.
// All GEMMs use v_wmma_f32_16x16x32_bf16 (f32 accumulate).
// ---------------------------------------------------------------------------

typedef __attribute__((ext_vector_type(16))) __bf16 v16bf;
typedef __attribute__((ext_vector_type(8)))  __bf16 v8bf;
typedef __attribute__((ext_vector_type(8)))  float  v8f;

union V16U { v16bf v; v8bf h[2]; };

#define GN_BS    512
#define GN_N     32
#define NTHREADS 256
#define NWAVES   8
#define HSTRIDE  72   // padded bf16 row stride for hidden staging tile

// ---------------- LDS layout (bytes) ----------------
constexpr int OFF_VFEAT = 0;                     // f32 [32][20]      = 2560
constexpr int OFF_VP0   = OFF_VFEAT + 2560;      // bf16 [32][64]     = 4096
constexpr int OFF_UP0B  = OFF_VP0   + 4096;      // bf16 [64]         = 128
constexpr int OFF_AGGEB = OFF_UP0B  + 128;       // bf16 [32][64]     = 4096
constexpr int OFF_UP0F  = OFF_AGGEB + 4096;      // f32 [64]          = 256
constexpr int OFF_UP1F  = OFF_UP0F  + 256;       // f32 [64]          = 256
constexpr int OFF_AGGE  = OFF_UP1F  + 256;       // f32 [32][64]      = 8192
constexpr int OFF_AGGEG = OFF_AGGE  + 8192;      // f32 [64]          = 256
constexpr int OFF_AGGVG = OFF_AGGEG + 256;       // f32 [64]          = 256
constexpr int OFF_GBUF  = OFF_AGGVG + 256;       // f32 [192]         = 768
constexpr int OFF_HG    = OFF_GBUF  + 768;       // f32 [64]          = 256
constexpr int OFF_HST   = OFF_HG    + 256;       // bf16 [8][16][72]  = 18432
constexpr int OFF_EP0   = OFF_HST   + 18432;     // bf16 [1024][64]   = 131072
// Swizzled bf16 weights: 4096 B per 32-wide K-step (4 ntiles * 32 lanes * 16 * 2B)
constexpr int OFF_W1E0  = OFF_EP0   + 131072;    // 2 ksteps  = 8192
constexpr int OFF_W2E0  = OFF_W1E0  + 8192;      // 2 ksteps  = 8192
constexpr int OFF_W1N0  = OFF_W2E0  + 8192;      // 3 ksteps  = 12288
constexpr int OFF_W2N0  = OFF_W1N0  + 12288;     // 2 ksteps  = 8192
constexpr int OFF_W1E1  = OFF_W2N0  + 8192;      // 8 ksteps  = 32768
constexpr int OFF_W2E1  = OFF_W1E1  + 32768;     // 2 ksteps  = 8192
constexpr int OFF_W1N1  = OFF_W2E1  + 8192;      // 6 ksteps  = 24576
constexpr int OFF_W2N1  = OFF_W1N1  + 24576;     // 2 ksteps  = 8192
constexpr int LDS_TOTAL = OFF_W2N1  + 8192;      // = 281216 B (< 320 KB/WGP)

// ---------------- helpers ----------------
// swish via fast hardware reciprocal (v_rcp_f32): 1+e^-x >= 1 so rcp is safe.
__device__ __forceinline__ float swishf(float x) {
  return x * __builtin_amdgcn_rcpf(1.f + __expf(-x));
}

// B-operand swizzle: element (k, n) of a [din x 64] weight goes to fragment
// (kstep = k/32, ntile = n/16), lane = (n&15)|(kh<<4) with kh = ((k&31)>>3)&1,
// element t = (k&7)|(((k&31)>>4)<<3).  Lane data is 16 contiguous bf16 -> 2x b128 LDS loads.
__device__ __forceinline__ int swz_idx(int k, int n) {
  int ks = k >> 5, kk = k & 31;
  int nt = n >> 4, nn = n & 15;
  int lane = nn | (((kk >> 3) & 1) << 4);
  return ((ks * 4 + nt) * 32 + lane) * 16 + ((kk & 7) | ((kk >> 4) << 3));
}

__device__ __forceinline__ void stage_w(__bf16* dst, const float* src, int din,
                                        int dinp, int tid) {
  for (int idx = tid; idx < dinp * 64; idx += NTHREADS) {
    int k = idx >> 6, n = idx & 63;
    float v = (k < din) ? src[k * 64 + n] : 0.f;
    dst[swz_idx(k, n)] = (__bf16)v;
  }
}

__device__ __forceinline__ v16bf load_bfrag(const __bf16* swz, int ks, int nt,
                                            int lane) {
  const v8bf* p = (const v8bf*)(swz + (((ks * 4 + nt) * 32 + lane) << 4));
  V16U u; u.h[0] = p[0]; u.h[1] = p[1];
  return u.v;
}

// 16-bit A fragment from a bf16 row with >=32 contiguous valid elements.
// ISA layout: elems 0-7 -> K = kh*8 + t ; elems 8-15 -> K = 16 + kh*8 + (t-8)
__device__ __forceinline__ v16bf load_afrag_row(const __bf16* rowk, int kh) {
  V16U u;
  u.h[0] = *(const v8bf*)(rowk + kh * 8);
  u.h[1] = *(const v8bf*)(rowk + 16 + kh * 8);
  return u.v;
}

// Generic element-wise A fragment builder (for ragged concat inputs).
template <typename FK>
__device__ __forceinline__ v16bf make_afrag(FK f, int kbase, int lane) {
  v16bf a;
  const int kh = lane >> 4;
#pragma unroll
  for (int t = 0; t < 16; ++t) {
    int k = kbase + ((t >> 3) << 4) + kh * 8 + (t & 7);
    a[t] = f(k);
  }
  return a;
}

// One 16-row tile through a 2-layer MLP (din -> 64 -> 64, swish both).
// afrag(ks) supplies the A fragment for K-step ks of the first GEMM.
// b1v/b2v are per-lane hoisted bias values (bias[nt*16 + (lane&15)]).
template <int NKS1, typename AF>
__device__ __forceinline__ void mlp_tile(AF&& afrag, const __bf16* w1s,
                                         const float b1v[4], const __bf16* w2s,
                                         const float b2v[4], __bf16* hst,
                                         int lane, v8f out[4]) {
  v8f acc[4];
#pragma unroll
  for (int nt = 0; nt < 4; ++nt)
#pragma unroll
    for (int r = 0; r < 8; ++r) acc[nt][r] = 0.f;

#pragma unroll
  for (int ks = 0; ks < NKS1; ++ks) {
    v16bf a = afrag(ks);
#pragma unroll
    for (int nt = 0; nt < 4; ++nt) {
      v16bf bb = load_bfrag(w1s, ks, nt, lane);
      acc[nt] = __builtin_amdgcn_wmma_f32_16x16x32_bf16(
          false, a, false, bb, (short)0, acc[nt], false, false);
    }
  }

  const int col = lane & 15, kh = lane >> 4;
  // bias + swish, stage hidden [16][64] (bf16) for D->A relayout via LDS
#pragma unroll
  for (int nt = 0; nt < 4; ++nt) {
#pragma unroll
    for (int r = 0; r < 8; ++r) {
      int row = r + 8 * kh;
      hst[row * HSTRIDE + nt * 16 + col] = (__bf16)swishf(acc[nt][r] + b1v[nt]);
    }
  }

  v8f acc2[4];
#pragma unroll
  for (int nt = 0; nt < 4; ++nt)
#pragma unroll
    for (int r = 0; r < 8; ++r) acc2[nt][r] = 0.f;

  const __bf16* hrow = hst + (lane & 15) * HSTRIDE;
#pragma unroll
  for (int ks = 0; ks < 2; ++ks) {
    v16bf a = load_afrag_row(hrow + ks * 32, kh);
#pragma unroll
    for (int nt = 0; nt < 4; ++nt) {
      v16bf bb = load_bfrag(w2s, ks, nt, lane);
      acc2[nt] = __builtin_amdgcn_wmma_f32_16x16x32_bf16(
          false, a, false, bb, (short)0, acc2[nt], false, false);
    }
  }
#pragma unroll
  for (int nt = 0; nt < 4; ++nt) {
#pragma unroll
    for (int r = 0; r < 8; ++r) out[nt][r] = swishf(acc2[nt][r] + b2v[nt]);
  }
}

// Tiny per-molecule global MLP (1 row) in f32, weights straight from global.
__device__ void glob_mlp(const float* gin, int din, const float* w1,
                         const float* b1, const float* w2, const float* b2,
                         float* hgl, float* uout, int tid) {
  if (tid < 64) {
    float s = b1[tid];
    for (int k = 0; k < din; ++k) s += gin[k] * w1[k * 64 + tid];
    hgl[tid] = swishf(s);
  }
  __syncthreads();
  if (tid < 64) {
    float s = b2[tid];
    for (int k = 0; k < 64; ++k) s += hgl[k] * w2[k * 64 + tid];
    uout[tid] = swishf(s);
  }
  __syncthreads();
}

struct GNArgs {
  const float *charges, *one_hot, *positions;
  const float *e_b1[2], *e_b2[2], *e_w1[2], *e_w2[2];
  const float *n_b1[2], *n_b2[2], *n_w1[2], *n_w2[2];
  const float *g_b1[2], *g_b2[2], *g_w1[2], *g_w2[2];
  const float *lin_w, *lin_b;
  float* out;
};

__global__ __launch_bounds__(NTHREADS, 1) void molec_gn_kernel(GNArgs A) {
  extern __shared__ char smem[];
  const int tid  = threadIdx.x;
  const int lane = tid & 31;
  const int wave = tid >> 5;
  const int b    = blockIdx.x;

  float*  vfeat  = (float*)(smem + OFF_VFEAT);   // [32][20] node feats (layer0)
  __bf16* vp0    = (__bf16*)(smem + OFF_VP0);    // [32][64] nodes after layer0
  __bf16* up0b   = (__bf16*)(smem + OFF_UP0B);   // [64]     u after layer0 (bf16)
  __bf16* aggeb  = (__bf16*)(smem + OFF_AGGEB);  // [32][64] agg_e1 (bf16)
  float*  up0f   = (float*)(smem + OFF_UP0F);
  float*  up1f   = (float*)(smem + OFF_UP1F);
  float*  agg_e  = (float*)(smem + OFF_AGGE);    // [32][64]
  float*  agg_eg = (float*)(smem + OFF_AGGEG);
  float*  agg_vg = (float*)(smem + OFF_AGGVG);
  float*  gbuf   = (float*)(smem + OFF_GBUF);    // [192]
  float*  hg     = (float*)(smem + OFF_HG);
  __bf16* hstage = (__bf16*)(smem + OFF_HST);    // [8][16][72]
  __bf16* ep0    = (__bf16*)(smem + OFF_EP0);    // [1024][64] layer0 edge feats
  __bf16* w1e0 = (__bf16*)(smem + OFF_W1E0);
  __bf16* w2e0 = (__bf16*)(smem + OFF_W2E0);
  __bf16* w1n0 = (__bf16*)(smem + OFF_W1N0);
  __bf16* w2n0 = (__bf16*)(smem + OFF_W2N0);
  __bf16* w1e1 = (__bf16*)(smem + OFF_W1E1);
  __bf16* w2e1 = (__bf16*)(smem + OFF_W2E1);
  __bf16* w1n1 = (__bf16*)(smem + OFF_W1N1);
  __bf16* w2n1 = (__bf16*)(smem + OFF_W2N1);
  __bf16* hst  = hstage + wave * 16 * HSTRIDE;

  // ---- Stage swizzled bf16 weights (L2-resident, per-block) ----
  stage_w(w1e0, A.e_w1[0], 38, 64, tid);
  stage_w(w2e0, A.e_w2[0], 64, 64, tid);
  stage_w(w1n0, A.n_w1[0], 83, 96, tid);
  stage_w(w2n0, A.n_w2[0], 64, 64, tid);
  stage_w(w1e1, A.e_w1[1], 256, 256, tid);
  stage_w(w2e1, A.e_w2[1], 64, 64, tid);
  stage_w(w1n1, A.n_w1[1], 192, 192, tid);
  stage_w(w2n1, A.n_w2[1], 64, 64, tid);

  // ---- Featurize: [onehot x {1,c,c^2} (15), pos (3), pad (2)] per node ----
  for (int idx = tid; idx < GN_N * 20; idx += NTHREADS) {
    int n = idx / 20, k = idx % 20;
    float val = 0.f;
    if (k < 15) {
      int s = k / 3, p = k % 3;
      float c = A.charges[b * GN_N + n] * (1.f / 9.f);
      float cp = (p == 0) ? 1.f : ((p == 1) ? c : c * c);
      val = A.one_hot[(b * GN_N + n) * 5 + s] * cp;
    } else if (k < 18) {
      val = A.positions[(b * GN_N + n) * 3 + (k - 15)];
    }
    vfeat[idx] = val;
  }
  for (int x = tid; x < GN_N * 64; x += NTHREADS) agg_e[x] = 0.f;
  if (tid < 64) agg_vg[tid] = 0.f;
  __syncthreads();

  const int col = lane & 15, khh = lane >> 4;

  // Hoist per-lane bias values for all WMMA MLPs (loop-invariant).
  float b1e0v[4], b2e0v[4], b1e1v[4], b2e1v[4];
#pragma unroll
  for (int nt = 0; nt < 4; ++nt) {
    b1e0v[nt] = A.e_b1[0][nt * 16 + col];
    b2e0v[nt] = A.e_b2[0][nt * 16 + col];
    b1e1v[nt] = A.e_b1[1][nt * 16 + col];
    b2e1v[nt] = A.e_b2[1][nt * 16 + col];
  }

  // ================= LAYER 0 =================
  // Edge MLP: input [v_j(18), v_i(18), e=1, u=1] padded to 64.
  for (int t = wave; t < 64; t += NWAVES) {
    const int i = t >> 1, jb = (t & 1) * 16;
    const int m = lane & 15;
    const int j = jb + m;
    auto af = [&](int ks) {
      return make_afrag(
          [&](int k) -> __bf16 {
            float v;
            if (k < 18)      v = vfeat[j * 20 + k];
            else if (k < 36) v = vfeat[i * 20 + (k - 18)];
            else if (k < 38) v = 1.f;
            else             v = 0.f;
            return (__bf16)v;
          },
          ks * 32, lane);
    };
    v8f ep[4];
    mlp_tile<2>(af, w1e0, b1e0v, w2e0, b2e0v, hst, lane, ep);
#pragma unroll
    for (int nt = 0; nt < 4; ++nt) {
      const int n = nt * 16 + col;
      float s = 0.f;
#pragma unroll
      for (int r = 0; r < 8; ++r) {
        const int row = r + 8 * khh;
        ep0[(i * 32 + jb + row) * 64 + n] = (__bf16)ep[nt][r];
        s += ep[nt][r];
      }
      atomicAdd(&agg_e[i * 64 + n], s);  // segment_sum over j (col == i)
    }
  }
  __syncthreads();
  if (tid < 64) {  // agg over all edges = sum_i agg_e[i]
    float s = 0.f;
    for (int i2 = 0; i2 < 32; ++i2) s += agg_e[i2 * 64 + tid];
    agg_eg[tid] = s;
  }
  __syncthreads();

  // Node MLP: input [v(18), agg_e(64), u=1] padded to 96.  2 tiles.
  if (wave < 2) {
    float b1nv[4], b2nv[4];
#pragma unroll
    for (int nt = 0; nt < 4; ++nt) {
      b1nv[nt] = A.n_b1[0][nt * 16 + col];
      b2nv[nt] = A.n_b2[0][nt * 16 + col];
    }
    const int nb = wave * 16;
    const int m = lane & 15;
    const int node = nb + m;
    auto af = [&](int ks) {
      return make_afrag(
          [&](int k) -> __bf16 {
            float v;
            if (k < 18)       v = vfeat[node * 20 + k];
            else if (k < 82)  v = agg_e[node * 64 + (k - 18)];
            else if (k == 82) v = 1.f;
            else              v = 0.f;
            return (__bf16)v;
          },
          ks * 32, lane);
    };
    v8f vpo[4];
    mlp_tile<3>(af, w1n0, b1nv, w2n0, b2nv, hst, lane, vpo);
#pragma unroll
    for (int nt = 0; nt < 4; ++nt) {
      const int n = nt * 16 + col;
      float s = 0.f;
#pragma unroll
      for (int r = 0; r < 8; ++r) {
        vp0[(nb + r + 8 * khh) * 64 + n] = (__bf16)vpo[nt][r];
        s += vpo[nt][r];
      }
      atomicAdd(&agg_vg[n], s);
    }
  }
  __syncthreads();

  // Global MLP 0: input [1, agg_eg(64), agg_vg(64)] = 129
  if (tid == 0) gbuf[0] = 1.f;
  if (tid < 64) { gbuf[1 + tid] = agg_eg[tid]; gbuf[65 + tid] = agg_vg[tid]; }
  __syncthreads();
  glob_mlp(gbuf, 129, A.g_w1[0], A.g_b1[0], A.g_w2[0], A.g_b2[0], hg, up0f, tid);

  // prep for layer 1
  for (int x = tid; x < GN_N * 64; x += NTHREADS) agg_e[x] = 0.f;
  if (tid < 64) up0b[tid] = (__bf16)up0f[tid];
  __syncthreads();

  // ================= LAYER 1 =================
  // Edge MLP: input [vp_j(64), vp_i(64), ep0(64), u(64)] = 256, all bf16 rows.
  for (int t = wave; t < 64; t += NWAVES) {
    const int i = t >> 1, jb = (t & 1) * 16;
    const int m = lane & 15;
    const int j = jb + m;
    const int e = i * 32 + j;
    auto af = [&](int ks) {
      const int chunk = ks >> 1, off = (ks & 1) * 32;
      const __bf16* rowp;
      if (chunk == 0)      rowp = vp0 + j * 64 + off;
      else if (chunk == 1) rowp = vp0 + i * 64 + off;
      else if (chunk == 2) rowp = ep0 + e * 64 + off;
      else                 rowp = up0b + off;
      return load_afrag_row(rowp, khh);
    };
    v8f ep[4];
    mlp_tile<8>(af, w1e1, b1e1v, w2e1, b2e1v, hst, lane, ep);
#pragma unroll
    for (int nt = 0; nt < 4; ++nt) {  // ep1 never stored; aggregate only
      const int n = nt * 16 + col;
      float s = 0.f;
#pragma unroll
      for (int r = 0; r < 8; ++r) s += ep[nt][r];
      atomicAdd(&agg_e[i * 64 + n], s);
    }
  }
  __syncthreads();
  if (tid < 64) {
    float s = 0.f;
    for (int i2 = 0; i2 < 32; ++i2) s += agg_e[i2 * 64 + tid];
    agg_eg[tid] = s;
  }
  for (int x = tid; x < GN_N * 64; x += NTHREADS) aggeb[x] = (__bf16)agg_e[x];
  if (tid < 64) agg_vg[tid] = 0.f;
  __syncthreads();

  // Node MLP 1: input [vp0(64), agg_e1(64), u(64)] = 192, bf16 fast path.
  if (wave < 2) {
    float b1nv[4], b2nv[4];
#pragma unroll
    for (int nt = 0; nt < 4; ++nt) {
      b1nv[nt] = A.n_b1[1][nt * 16 + col];
      b2nv[nt] = A.n_b2[1][nt * 16 + col];
    }
    const int nb = wave * 16;
    const int m = lane & 15;
    const int node = nb + m;
    auto af = [&](int ks) {
      const int chunk = ks >> 1, off = (ks & 1) * 32;
      const __bf16* rowp;
      if (chunk == 0)      rowp = vp0 + node * 64 + off;
      else if (chunk == 1) rowp = aggeb + node * 64 + off;
      else                 rowp = up0b + off;
      return load_afrag_row(rowp, khh);
    };
    v8f vpo[4];
    mlp_tile<6>(af, w1n1, b1nv, w2n1, b2nv, hst, lane, vpo);
#pragma unroll
    for (int nt = 0; nt < 4; ++nt) {  // vp1 only needed for agg_vg
      const int n = nt * 16 + col;
      float s = 0.f;
#pragma unroll
      for (int r = 0; r < 8; ++r) s += vpo[nt][r];
      atomicAdd(&agg_vg[n], s);
    }
  }
  __syncthreads();

  // Global MLP 1: input [u(64), agg_eg(64), agg_vg(64)] = 192
  if (tid < 64) {
    gbuf[tid] = up0f[tid];
    gbuf[64 + tid] = agg_eg[tid];
    gbuf[128 + tid] = agg_vg[tid];
  }
  __syncthreads();
  glob_mlp(gbuf, 192, A.g_w1[1], A.g_b1[1], A.g_w2[1], A.g_b2[1], hg, up1f, tid);

  // Final linear readout
  if (tid == 0) {
    float s = A.lin_b[0];
    for (int k = 0; k < 64; ++k) s += up1f[k] * A.lin_w[k];
    A.out[b] = s;
  }
}

extern "C" void kernel_launch(void* const* d_in, const int* in_sizes, int n_in,
                              void* d_out, int out_size, void* d_ws, size_t ws_size,
                              hipStream_t stream) {
  (void)d_ws; (void)ws_size; (void)out_size;
  GNArgs A;
  A.charges   = (const float*)d_in[0];
  A.one_hot   = (const float*)d_in[1];
  A.positions = (const float*)d_in[2];

  // Detect pytree flatten order from element counts.
  // Fields within an MLP: sorted keys -> (b1,b2,w1,w2); insertion -> (w1,b1,w2,b2)
  int F_B1, F_B2, F_W1, F_W2;
  if (in_sizes[3] <= 64) { F_B1 = 0; F_B2 = 1; F_W1 = 2; F_W2 = 3; }
  else                   { F_W1 = 0; F_B1 = 1; F_W2 = 2; F_B2 = 3; }
  // Groups in a layer: sorted -> (edge, glob, node); insertion -> (edge, node, glob).
  // Layer-0 w1 sizes: edge 38*64=2432, glob 129*64=8256, node 83*64=5312.
  int g2w1 = in_sizes[3 + 4 + F_W1];
  int G_EDGE = 0, G_GLOB, G_NODE;
  if (g2w1 == 129 * 64) { G_GLOB = 1; G_NODE = 2; }
  else                  { G_NODE = 1; G_GLOB = 2; }

  for (int l = 0; l < 2; ++l) {
    int base = 3 + l * 12;
    A.e_b1[l] = (const float*)d_in[base + G_EDGE * 4 + F_B1];
    A.e_b2[l] = (const float*)d_in[base + G_EDGE * 4 + F_B2];
    A.e_w1[l] = (const float*)d_in[base + G_EDGE * 4 + F_W1];
    A.e_w2[l] = (const float*)d_in[base + G_EDGE * 4 + F_W2];
    A.g_b1[l] = (const float*)d_in[base + G_GLOB * 4 + F_B1];
    A.g_b2[l] = (const float*)d_in[base + G_GLOB * 4 + F_B2];
    A.g_w1[l] = (const float*)d_in[base + G_GLOB * 4 + F_W1];
    A.g_w2[l] = (const float*)d_in[base + G_GLOB * 4 + F_W2];
    A.n_b1[l] = (const float*)d_in[base + G_NODE * 4 + F_B1];
    A.n_b2[l] = (const float*)d_in[base + G_NODE * 4 + F_B2];
    A.n_w1[l] = (const float*)d_in[base + G_NODE * 4 + F_W1];
    A.n_w2[l] = (const float*)d_in[base + G_NODE * 4 + F_W2];
  }
  // top level: sorted -> (layers, lin_b, lin_w); insertion -> (layers, lin_w, lin_b)
  if (n_in >= 29 && in_sizes[27] == 1) {
    A.lin_b = (const float*)d_in[27];
    A.lin_w = (const float*)d_in[28];
  } else {
    A.lin_w = (const float*)d_in[27];
    A.lin_b = (const float*)d_in[28];
  }
  A.out = (float*)d_out;

  (void)hipFuncSetAttribute((const void*)molec_gn_kernel,
                            hipFuncAttributeMaxDynamicSharedMemorySize,
                            LDS_TOTAL);
  molec_gn_kernel<<<GN_BS, NTHREADS, LDS_TOTAL, stream>>>(A);
}
